// CrossTransformerBlock_24292335026198
// MI455X (gfx1250) — matrix operand order repaired and checked by reading, not compiled, not run on hardware
//
#include <hip/hip_runtime.h>
#include <hip/hip_bf16.h>

typedef __attribute__((ext_vector_type(16))) _Float16 v16h;
typedef __attribute__((ext_vector_type(8)))  float    v8f;
typedef int v4i_vs __attribute__((vector_size(16)));   // matches builtin param type

#define N_EMBD 512
#define NH     8
#define HD     64
#define TOK    1024      // H*W = 32*32
#define BATCH  8
#define MTOT   (BATCH * TOK)   // 8192 tokens total

// ---------------------------------------------------------------------------
// CDNA5 async global->LDS path (ASYNCcnt). Signature per hipcc diagnostic:
// param0 = global (as1) pointer to int4, param1 = LDS (as3) pointer to int4,
// then two i32 immediates (offset, cpol).
// ---------------------------------------------------------------------------
#if defined(__has_builtin)
#  if __has_builtin(__builtin_amdgcn_global_load_async_to_lds_b128)
#    define HAVE_ASYNC_LDS 1
#  endif
#endif
#ifndef HAVE_ASYNC_LDS
#  define HAVE_ASYNC_LDS 0
#endif

#if HAVE_ASYNC_LDS
#  if __has_builtin(__builtin_amdgcn_s_wait_asynccnt)
#    define WAIT_ASYNC(n) __builtin_amdgcn_s_wait_asynccnt(n)
#  else
#    define WAIT_ASYNC(n) asm volatile("s_wait_asynccnt %0" ::"i"(n) : "memory")
#  endif
// Copy 16 halfs (32B) global -> LDS as two async B128 transfers.
__device__ __forceinline__ void async_copy16h(const _Float16* g, _Float16* l) {
  auto* gp = (__attribute__((address_space(1))) v4i_vs*)g;
  auto* lp = (__attribute__((address_space(3))) v4i_vs*)l;
  __builtin_amdgcn_global_load_async_to_lds_b128(gp, lp, 0, 0);
  __builtin_amdgcn_global_load_async_to_lds_b128(gp + 1, lp + 1, 0, 0);
}
#else
#  define WAIT_ASYNC(n)
__device__ __forceinline__ void async_copy16h(const _Float16* g, _Float16* l) {
  uint4* d = reinterpret_cast<uint4*>(l);
  const uint4* s = reinterpret_cast<const uint4*>(g);
  d[0] = s[0];
  d[1] = s[1];
}
#endif

// ---------------------------------------------------------------------------
// WMMA fragment helpers (CDNA5 ISA §7.12.2, wave32 layouts)
// ---------------------------------------------------------------------------
__device__ __forceinline__ int lane_id() { return (int)(threadIdx.x & 31u); }

__device__ __forceinline__ v8f vzero8() {
  v8f z;
#pragma unroll
  for (int r = 0; r < 8; ++r) z[r] = 0.0f;
  return z;
}

// A-matrix 16x32 f16 fragment; src = tile origin, ld = row stride (halfs).
// Lanes 0-15: row=lane, halfs K[0..7],K[16..23]; lanes 16-31: K[8..15],K[24..31].
__device__ __forceinline__ v16h load_a16x32(const _Float16* __restrict__ src, int ld) {
  const int lane = lane_id();
  const int row  = lane & 15;
  const int koff = (lane & 16) ? 8 : 0;
  const _Float16* p = src + (size_t)row * ld + koff;
  v16h r;
  reinterpret_cast<uint4*>(&r)[0] = *reinterpret_cast<const uint4*>(p);
  reinterpret_cast<uint4*>(&r)[1] = *reinterpret_cast<const uint4*>(p + 16);
  return r;
}

// B-matrix 32x16 f16 fragment from column-major view src[col*ld + k].
// Lanes 0-15: col=n0+lane, K[0..7],K[16..23] ; lanes 16-31: col=n0+lane-16, K[8..15],K[24..31].
__device__ __forceinline__ v16h load_b32x16(const _Float16* __restrict__ src, int ld, int n0) {
  const int lane = lane_id();
  const int col  = n0 + (lane & 15);
  const int k    = (lane & 16) ? 8 : 0;
  const _Float16* p = src + (size_t)col * ld + k;
  v16h r;
  reinterpret_cast<uint4*>(&r)[0] = *reinterpret_cast<const uint4*>(p);
  reinterpret_cast<uint4*>(&r)[1] = *reinterpret_cast<const uint4*>(p + 16);
  return r;
}

__device__ __forceinline__ v8f wmma_f16(v16h a, v16h b, v8f c) {
  return __builtin_amdgcn_wmma_f32_16x16x32_f16(false, a, false, b, (short)0, c,
                                                false, false);
}

// ---------------------------------------------------------------------------
// K0: weights f32 -> f16, TRANSPOSED to [n][k] so GEMM B-tiles are
// per-column contiguous (B-fragment loads become b128, no LDS transpose).
// ---------------------------------------------------------------------------
__global__ __launch_bounds__(256) void cvt_weights_t(
    const float* __restrict__ w0, const float* __restrict__ w1,
    const float* __restrict__ w2, const float* __restrict__ w3,
    _Float16* __restrict__ o) {
  const int i  = blockIdx.x * 256 + threadIdx.x;   // coalesced read index k*512+n
  const int NW = N_EMBD * N_EMBD;
  const int k  = i >> 9;
  const int n  = i & (N_EMBD - 1);
  const int ti = n * N_EMBD + k;                   // transposed write
  o[ti]          = (_Float16)w0[i];
  o[ti + NW]     = (_Float16)w1[i];
  o[ti + 2 * NW] = (_Float16)w2[i];
  o[ti + 3 * NW] = (_Float16)w3[i];
}

// ---------------------------------------------------------------------------
// K1: [B,C,H,W] -> token-major LayerNorm, f16 out [B*T, C]
// ---------------------------------------------------------------------------
__global__ __launch_bounds__(256) void ln_tokens(
    const float* __restrict__ x, const float* __restrict__ g,
    const float* __restrict__ b, _Float16* __restrict__ out) {
  const int token = blockIdx.x;        // b*TOK + t
  const int bb = token >> 10;
  const int t  = token & (TOK - 1);
  const float* xp = x + (size_t)bb * N_EMBD * TOK + t;
  const int c0 = threadIdx.x, c1 = threadIdx.x + 256;
  const float v0 = xp[(size_t)c0 * TOK];
  const float v1 = xp[(size_t)c1 * TOK];

  __shared__ float red[256];
  red[threadIdx.x] = v0 + v1;
  __syncthreads();
  for (int off = 128; off > 0; off >>= 1) {
    if ((int)threadIdx.x < off) red[threadIdx.x] += red[threadIdx.x + off];
    __syncthreads();
  }
  const float mu = red[0] * (1.0f / N_EMBD);
  __syncthreads();
  const float d0 = v0 - mu, d1 = v1 - mu;
  red[threadIdx.x] = d0 * d0 + d1 * d1;
  __syncthreads();
  for (int off = 128; off > 0; off >>= 1) {
    if ((int)threadIdx.x < off) red[threadIdx.x] += red[threadIdx.x + off];
    __syncthreads();
  }
  const float rstd = rsqrtf(red[0] * (1.0f / N_EMBD) + 1e-5f);
  _Float16* op = out + (size_t)token * N_EMBD;
  op[c0] = (_Float16)(d0 * rstd * g[c0] + b[c0]);
  op[c1] = (_Float16)(d1 * rstd * g[c1] + b[c1]);
}

// ---------------------------------------------------------------------------
// K2/K4: 8192x512 @ 512x512 GEMM. 128x64 tile per 128-thread block; each wave
// owns 32 rows x 64 cols (8 WMMA / k-step). Double-buffered async LDS staging.
// mode 0: f16 per-head out [B,NH,T,HD] (+bias)
// mode 1: f32 out [B,C,T] (+bias +residual)
// ---------------------------------------------------------------------------
__global__ __launch_bounds__(128) void gemm512(
    const _Float16* __restrict__ X,    // [MTOT, 512] row-major f16
    const _Float16* __restrict__ Wt,   // [512(n), 512(k)] TRANSPOSED f16
    const float* __restrict__ bias,    // [512]
    _Float16* __restrict__ outH,       // mode 0
    float* __restrict__ outF,          // mode 1
    const float* __restrict__ resid,   // mode 1: [B, C, T]
    int mode) {
  __shared__ __align__(16) _Float16 As[2][128 * 32];  // [row][k]
  __shared__ __align__(16) _Float16 Bs[2][64 * 32];   // [n][k]
  const int tid  = threadIdx.x;
  const int w    = tid >> 5;
  const int lane = tid & 31;
  const int m0   = blockIdx.y * 128;
  const int n0   = blockIdx.x * 64;

  v8f acc[2][4];
#pragma unroll
  for (int s = 0; s < 2; ++s)
#pragma unroll
    for (int i = 0; i < 4; ++i) acc[s][i] = vzero8();

  // 6 async B128 ops per thread per tile (A: 4, B: 2)
  auto issue_tile = [&](int kb, int buf) {
    const _Float16* ga = X + (size_t)(m0 + tid) * N_EMBD + kb;  // one full A row
    _Float16* la = &As[buf][tid * 32];
    async_copy16h(ga, la);
    async_copy16h(ga + 16, la + 16);
    const int col = tid >> 1, kc = (tid & 1) * 16;
    async_copy16h(Wt + (size_t)(n0 + col) * N_EMBD + kb + kc,
                  &Bs[buf][col * 32 + kc]);
  };

  issue_tile(0, 0);
  int buf = 0;
  for (int kb = 0; kb < N_EMBD; kb += 32) {
    if (kb + 32 < N_EMBD) {
      issue_tile(kb + 32, buf ^ 1);  // prefetch next tile into other buffer
      WAIT_ASYNC(6);                 // allow the 6 just-issued ops to stay in flight
    } else {
      WAIT_ASYNC(0);
    }
    __syncthreads();

    const v16h a0 = load_a16x32(&As[buf][(32 * w) * 32], 32);
    const v16h a1 = load_a16x32(&As[buf][(32 * w + 16) * 32], 32);
#pragma unroll
    for (int nc = 0; nc < 4; ++nc) {
      const v16h bf = load_b32x16(Bs[buf], 32, nc * 16);
      acc[0][nc] = wmma_f16(a0, bf, acc[0][nc]);
      acc[1][nc] = wmma_f16(a1, bf, acc[1][nc]);
    }
    __syncthreads();
    buf ^= 1;
  }

  // Epilogue: D layout -> lane = column, VGPR r = row (+8 for lanes 16-31)
  const int rofs = (lane & 16) ? 8 : 0;
#pragma unroll
  for (int si = 0; si < 2; ++si) {
    const int mbase = m0 + 32 * w + 16 * si + rofs;  // rows mbase..mbase+7
    const int bb    = mbase >> 10;                   // tile never crosses batch
    const int tbase = mbase & (TOK - 1);
#pragma unroll
    for (int nc = 0; nc < 4; ++nc) {
      const int   c  = n0 + nc * 16 + (lane & 15);
      const float bv = bias[c];
      if (mode == 0) {
        const int h = c >> 6, d = c & (HD - 1);
        _Float16* op = outH + ((size_t)(bb * NH + h) * TOK) * HD + d;
#pragma unroll
        for (int r = 0; r < 8; ++r)
          op[(size_t)(tbase + r) * HD] = (_Float16)(acc[si][nc][r] + bv);
      } else {
        const size_t base = ((size_t)bb * N_EMBD + c) * TOK + tbase;
#pragma unroll
        for (int r = 0; r < 8; ++r)
          outF[base + r] = acc[si][nc][r] + bv + resid[base + r];
      }
    }
  }
}

// ---------------------------------------------------------------------------
// K3: flash-style attention. Block = (64 query rows, one batch-head), 4 waves,
// each owning 16 query rows. K block staged via async-to-LDS (shared by all
// waves, no transpose needed); V block manually transposed into LDS.
// ---------------------------------------------------------------------------
__global__ __launch_bounds__(128) void attention64(
    const _Float16* __restrict__ Qm, const _Float16* __restrict__ Km,
    const _Float16* __restrict__ Vm, _Float16* __restrict__ Y) {
  __shared__ __align__(16) _Float16 Ks[32 * 64];      // [key][hd] (as in global)
  __shared__ __align__(16) _Float16 Vt[64 * 32];      // transposed [hd][key]
  __shared__ __align__(16) _Float16 Pb[4][16 * 32];   // per-wave P tile [row][key]
  const int tid  = threadIdx.x;
  const int w    = tid >> 5;
  const int lane = tid & 31;
  const int bh   = blockIdx.y;
  const int q0   = blockIdx.x * 64 + 16 * w;
  const _Float16* Qh = Qm + (size_t)bh * TOK * HD;
  const _Float16* Kh = Km + (size_t)bh * TOK * HD;
  const _Float16* Vh = Vm + (size_t)bh * TOK * HD;

  // Q rows resident as two 16x32 A-fragments (hd 0..31 / 32..63)
  const v16h qa0 = load_a16x32(Qh + (size_t)q0 * HD, HD);
  const v16h qa1 = load_a16x32(Qh + (size_t)q0 * HD + 32, HD);

  v8f accv[4];
#pragma unroll
  for (int i = 0; i < 4; ++i) accv[i] = vzero8();
  v8f mrow, lrow;
#pragma unroll
  for (int r = 0; r < 8; ++r) { mrow[r] = -1e30f; lrow[r] = 0.0f; }

  const float scale = 0.125f;  // 1/sqrt(64)

  for (int k0 = 0; k0 < TOK; k0 += 32) {
    {  // K block -> LDS, layout preserved (async, 2 B128 per thread)
      const int kr = tid >> 2, c0 = (tid & 3) * 16;
      async_copy16h(Kh + (size_t)(k0 + kr) * HD + c0, &Ks[kr * 64 + c0]);
    }
    {  // V block -> LDS transposed: Vt[hd][key]
      const int kr  = tid >> 2;
      const int nc0 = (tid & 3) * 16;
      v16h tmp;
      const uint4* src =
          reinterpret_cast<const uint4*>(Vh + (size_t)(k0 + kr) * HD + nc0);
      reinterpret_cast<uint4*>(&tmp)[0] = src[0];
      reinterpret_cast<uint4*>(&tmp)[1] = src[1];
#pragma unroll
      for (int j = 0; j < 16; ++j) Vt[(nc0 + j) * 32 + kr] = tmp[j];
    }
    WAIT_ASYNC(0);
    __syncthreads();

    // S = Q K^T (K-dim = hd = 64 -> 2 WMMAs per 16-key chunk)
    v8f s0 = vzero8(), s1 = vzero8();
    s0 = wmma_f16(qa0, load_b32x16(Ks, HD, 0), s0);
    s0 = wmma_f16(qa1, load_b32x16(Ks + 32, HD, 0), s0);
    s1 = wmma_f16(qa0, load_b32x16(Ks, HD, 16), s1);
    s1 = wmma_f16(qa1, load_b32x16(Ks + 32, HD, 16), s1);
#pragma unroll
    for (int r = 0; r < 8; ++r) { s0[r] *= scale; s1[r] *= scale; }

    // Row max across the 16 lanes of each half-wave (D layout: lane = key col)
    v8f cm;
#pragma unroll
    for (int r = 0; r < 8; ++r) cm[r] = fmaxf(s0[r], s1[r]);
#pragma unroll
    for (int m = 1; m < 16; m <<= 1)
#pragma unroll
      for (int r = 0; r < 8; ++r) cm[r] = fmaxf(cm[r], __shfl_xor(cm[r], m, 32));

    v8f mnew, alpha, p0, p1, rs;
#pragma unroll
    for (int r = 0; r < 8; ++r) {
      mnew[r]  = fmaxf(mrow[r], cm[r]);
      alpha[r] = __expf(mrow[r] - mnew[r]);
      p0[r]    = __expf(s0[r] - mnew[r]);
      p1[r]    = __expf(s1[r] - mnew[r]);
      rs[r]    = p0[r] + p1[r];
    }
#pragma unroll
    for (int m = 1; m < 16; m <<= 1)
#pragma unroll
      for (int r = 0; r < 8; ++r) rs[r] += __shfl_xor(rs[r], m, 32);
#pragma unroll
    for (int r = 0; r < 8; ++r) lrow[r] = lrow[r] * alpha[r] + rs[r];
#pragma unroll
    for (int nc = 0; nc < 4; ++nc)
#pragma unroll
      for (int r = 0; r < 8; ++r) accv[nc][r] *= alpha[r];
    mrow = mnew;

    // Spill P (D layout) to LDS, reload as 16x32 A-fragment.
    // Same-wave LDS ops are in-order: no barrier needed.
    _Float16* Pw  = Pb[w];
    const int rof = (lane & 16) ? 8 : 0;
    const int cl  = lane & 15;
#pragma unroll
    for (int r = 0; r < 8; ++r) {
      Pw[(rof + r) * 32 + cl]      = (_Float16)p0[r];
      Pw[(rof + r) * 32 + 16 + cl] = (_Float16)p1[r];
    }
    const v16h pa = load_a16x32(Pw, 32);
#pragma unroll
    for (int nc = 0; nc < 4; ++nc)
      accv[nc] = wmma_f16(pa, load_b32x16(Vt, 32, nc * 16), accv[nc]);

    __syncthreads();  // protect Ks/Vt before next block's overwrite
  }

  // Epilogue: y = acc / l  -> Y[b, t, c] f16, c = h*64 + d
  const int rof = (lane & 16) ? 8 : 0;
  const int cl  = lane & 15;
  const int b   = bh >> 3, h = bh & 7;
#pragma unroll
  for (int nc = 0; nc < 4; ++nc) {
    const int c = h * HD + nc * 16 + cl;
#pragma unroll
    for (int r = 0; r < 8; ++r) {
      const int t = q0 + rof + r;
      Y[((size_t)b * TOK + t) * N_EMBD + c] = (_Float16)(accv[nc][r] / lrow[r]);
    }
  }
}

// ---------------------------------------------------------------------------
// Launch
// ---------------------------------------------------------------------------
extern "C" void kernel_launch(void* const* d_in, const int* in_sizes, int n_in,
                              void* d_out, int out_size, void* d_ws, size_t ws_size,
                              hipStream_t stream) {
  (void)in_sizes; (void)n_in; (void)out_size; (void)ws_size;
  const float* q       = (const float*)d_in[0];
  const float* kv      = (const float*)d_in[1];
  const float* ln_kv_w = (const float*)d_in[2];
  const float* ln_kv_b = (const float*)d_in[3];
  const float* ln_q_w  = (const float*)d_in[4];
  const float* ln_q_b  = (const float*)d_in[5];
  const float* Wk = (const float*)d_in[6];
  const float* bk = (const float*)d_in[7];
  const float* Wq = (const float*)d_in[8];
  const float* bq = (const float*)d_in[9];
  const float* Wv = (const float*)d_in[10];
  const float* bv = (const float*)d_in[11];
  const float* Wp = (const float*)d_in[12];
  const float* bp = (const float*)d_in[13];
  float* out = (float*)d_out;

  _Float16* ws = (_Float16*)d_ws;
  const size_t NW = (size_t)N_EMBD * N_EMBD;   // 262144 halfs per weight
  const size_t NX = (size_t)MTOT * N_EMBD;     // 4194304 halfs per activation
  _Float16* w16 = ws;              // 4 transposed f16 weights
  _Float16* Xkv = ws + 4 * NW;     // LN(kv) tokens [MTOT, C]
  _Float16* Xq  = Xkv + NX;        // LN(q) tokens
  _Float16* Kb  = Xq + NX;         // K  [B,NH,T,HD]
  _Float16* Qb  = Kb + NX;         // Q  [B,NH,T,HD]
  _Float16* Vb  = Qb + NX;         // V  [B,NH,T,HD]
  _Float16* Yb  = Vb + NX;         // attn out [B,T,C]

  cvt_weights_t<<<(int)(NW / 256), 256, 0, stream>>>(Wk, Wq, Wv, Wp, w16);
  ln_tokens<<<MTOT, 256, 0, stream>>>(kv, ln_kv_w, ln_kv_b, Xkv);
  ln_tokens<<<MTOT, 256, 0, stream>>>(q,  ln_q_w,  ln_q_b,  Xq);

  dim3 ggrid(N_EMBD / 64, MTOT / 128);  // (8, 64)
  gemm512<<<ggrid, 128, 0, stream>>>(Xkv, w16 + 0 * NW, bk, Kb, nullptr, nullptr, 0);
  gemm512<<<ggrid, 128, 0, stream>>>(Xq,  w16 + 1 * NW, bq, Qb, nullptr, nullptr, 0);
  gemm512<<<ggrid, 128, 0, stream>>>(Xkv, w16 + 2 * NW, bv, Vb, nullptr, nullptr, 0);

  attention64<<<dim3(TOK / 64, BATCH * NH), 128, 0, stream>>>(Qb, Kb, Vb, Yb);

  gemm512<<<ggrid, 128, 0, stream>>>(Yb, w16 + 3 * NW, bp, nullptr, out, kv, 1);
}
